// Self_Attention_90950227460850
// MI455X (gfx1250) — matrix operand order repaired and checked by reading, not compiled, not run on hardware
//
#include <hip/hip_runtime.h>
#include <hip/hip_bf16.h>

typedef __attribute__((ext_vector_type(16))) _Float16 v16h;
typedef __attribute__((ext_vector_type(8)))  _Float16 v8h;
typedef __attribute__((ext_vector_type(8)))  float    v8f;

constexpr int Hc = 512, Wc = 512, Cc = 32, Bn = 2;
constexpr int RS = 40;   // LDS row stride in halves (80 B): 16 B-aligned, bank-conflict-free

// ---------------------------------------------------------------------------
// WMMA orientation: D(16x16) = A(16x32) x B(32x16),  A = weight tile (rows =
// out-channels), B = x^T (cols = voxels).  C-layout => lane = voxel, VGPR r =
// channel (8 consecutive) -> fully vectorized row-major global stores.
// ---------------------------------------------------------------------------

// Pre-packed A fragment (ISA 7.12.2 16-bit A layout), 32 B per lane, coalesced.
__device__ __forceinline__ v16h load_frag(const _Float16* frag, int tile, int lane) {
  const v8h* p = (const v8h*)(frag + ((size_t)tile * 32 + lane) * 16);
  v8h lo = p[0], hi = p[1];
  v16h a;
#pragma unroll
  for (int i = 0; i < 8; ++i) { a[i] = lo[i]; a[8 + i] = hi[i]; }
  return a;
}

// B = x^T fragment from LDS tile [16][RS] (row-major activations):
// lanes 0-15: col m=lane, K=0..7,16..23 ; lanes 16-31: col m=lane-16, K=8..15,24..31
__device__ __forceinline__ v16h load_actB(const _Float16* sh, int lane) {
  const v8h* p = (const v8h*)(sh + (lane & 15) * RS);
  v8h lo = p[lane >> 4], hi = p[(lane >> 4) + 2];
  v16h b;
#pragma unroll
  for (int i = 0; i < 8; ++i) { b[i] = lo[i]; b[8 + i] = hi[i]; }
  return b;
}

// ---- kernel 0: zero canvas, grid = -1 --------------------------------------
__global__ void init_kernel(float* canvas, long canvas4, int* grid, int gridN) {
  long i = (long)blockIdx.x * blockDim.x + threadIdx.x;
  long stride = (long)gridDim.x * blockDim.x;
  float4 z = make_float4(0.f, 0.f, 0.f, 0.f);
  for (long j = i; j < canvas4; j += stride) ((float4*)canvas)[j] = z;
  for (long j = i; j < gridN;   j += stride) grid[j] = -1;
}

// ---- kernel 1: scatter voxel indices into dense grid -----------------------
__global__ void scatter_grid_kernel(const int* coors, int* grid, int N) {
  long i = (long)blockIdx.x * blockDim.x + threadIdx.x;
  if (i >= (long)Bn * N) return;
  int b = (int)(i / N), v = (int)(i % N);
  grid[(size_t)b * Hc * Wc + coors[i * 2] * Wc + coors[i * 2 + 1]] = v;
}

// ---- kernel 2: prep — collapse weight chains, pack f16 A-fragments ---------
// Weff_m = in_w[m] @ W1_m (32x32), beff_m = in_w[m] @ b1_m + in_b[m]
// pos2[s] = (SHIFT[s] @ pos_w^T + pos_b) @ in_wv^T
// frag tiles: 0/1 = Weff_q, 2/3 = Weff_k, 4/5 = Weff_v, 6/7 = out_w  (nh = n0/16)
__global__ void prep_kernel(const float* wq, const float* bq,
                            const float* wk, const float* bk,
                            const float* wv, const float* bv,
                            const float* in_w, const float* in_b,
                            const float* out_w, const float* pos_w, const float* pos_b,
                            _Float16* frag, float* beff, float* pos2) {
  __shared__ float Weff[3][32][32];
  int t = threadIdx.x;                       // 1024 threads
  int n = t >> 5, k = t & 31;
  const float* W1[3] = {wq, wk, wv};
#pragma unroll
  for (int mm = 0; mm < 3; ++mm) {
    const float* w2r = in_w + mm * 1024 + n * 32;
    float acc = 0.f;
    for (int j = 0; j < 32; ++j) acc = fmaf(w2r[j], W1[mm][j * 32 + k], acc);
    Weff[mm][n][k] = acc;
  }
  if (t < 96) {                              // effective biases
    int mm = t >> 5, nn = t & 31;
    const float* b1 = (mm == 0) ? bq : ((mm == 1) ? bk : bv);
    const float* w2r = in_w + mm * 1024 + nn * 32;
    float acc = in_b[mm * 32 + nn];
    for (int j = 0; j < 32; ++j) acc = fmaf(w2r[j], b1[j], acc);
    beff[t] = acc;
  }
  if (t < 288) {                             // projected position table
    const int sh0[9] = {0,-1,1,0,-1,1,0,-1, 1};
    const int sh1[9] = {0, 0,0,1, 1,1,-1,-1,-1};
    int s = t >> 5, c = t & 31;
    const float* wv2 = in_w + (64 + c) * 32;
    float acc = 0.f;
    for (int cc = 0; cc < 32; ++cc) {
      float pv = fmaf((float)sh0[s], pos_w[cc * 2],
                 fmaf((float)sh1[s], pos_w[cc * 2 + 1], pos_b[cc]));
      acc = fmaf(pv, wv2[cc], acc);
    }
    pos2[t] = acc;
  }
  __syncthreads();
  if (t < 256) {                             // pack A fragments (f16)
    int tile = t >> 5, lane = t & 31;
    int nh  = tile & 1;
    int row = nh * 16 + (lane & 15);
    int kb  = (lane >> 4) * 8;
    const float* src = (tile < 6) ? &Weff[tile >> 1][row][0] : (out_w + row * 32);
    _Float16* d = frag + (size_t)t * 16;
#pragma unroll
    for (int i = 0; i < 8; ++i) { d[i] = (_Float16)src[kb + i]; d[8 + i] = (_Float16)src[kb + 16 + i]; }
  }
}

// ---- kernel 3 (pass 1): LN + fused q/k/v projection, 6 WMMA / 16-voxel tile -
__global__ void pass1_kernel(const float* feats, const float* ln_w, const float* ln_b,
                             const _Float16* frag, const float* beff,
                             _Float16* Q2, _Float16* K2, _Float16* V2, int N) {
  __shared__ _Float16 xh[16 * RS];
  int lane = threadIdx.x;
  int bidx = blockIdx.y;
  long base = (long)blockIdx.x * 16;

  if (lane < 16) {                           // one voxel per lane: LayerNorm
    long vox = base + lane; if (vox >= N) vox = N - 1;
    const float* f = feats + ((size_t)bidx * N + vox) * Cc;
    float x[Cc];
#pragma unroll
    for (int k = 0; k < Cc; ++k) x[k] = f[k];
    float sum = 0.f;
#pragma unroll
    for (int k = 0; k < Cc; ++k) sum += x[k];
    float mu = sum * (1.f / Cc);
    float sq = 0.f;
#pragma unroll
    for (int k = 0; k < Cc; ++k) { float d = x[k] - mu; sq = fmaf(d, d, sq); }
    float rs = rsqrtf(sq * (1.f / Cc) + 1e-5f);
    v8h row[4];
#pragma unroll
    for (int k = 0; k < Cc; ++k)
      row[k >> 3][k & 7] = (_Float16)fmaf((x[k] - mu) * rs, ln_w[k], ln_b[k]);
    v8h* dst = (v8h*)(xh + lane * RS);
#pragma unroll
    for (int i = 0; i < 4; ++i) dst[i] = row[i];   // 4x ds_store_b128
  }
  __syncthreads();

  v16h b = load_actB(xh, lane);
  int hi = lane >> 4, lo = lane & 15;
  long vox = base + lo;
  bool ok = vox < N;
  _Float16* outs[3] = {Q2 + (size_t)bidx * N * Cc,
                       K2 + (size_t)bidx * N * Cc,
                       V2 + (size_t)bidx * N * Cc};
#pragma unroll
  for (int mm = 0; mm < 3; ++mm) {
#pragma unroll
    for (int nh = 0; nh < 2; ++nh) {
      v16h a = load_frag(frag, mm * 2 + nh, lane);
      const float* bb = beff + mm * 32 + nh * 16 + 8 * hi;
      v8f c;
#pragma unroll
      for (int r = 0; r < 8; ++r) c[r] = bb[r];   // bias preload
      c = __builtin_amdgcn_wmma_f32_16x16x32_f16(false, a, false, b, (short)0, c, false, false);
      if (ok) {
        v8h o;
#pragma unroll
        for (int r = 0; r < 8; ++r) o[r] = (_Float16)c[r];
        *(v8h*)(outs[mm] + vox * Cc + nh * 16 + 8 * hi) = o;  // 1x global_store_b128
      }
    }
  }
}

// ---- kernel 4 (pass 2): gather + softmax + out-proj WMMA + canvas scatter ---
__global__ void pass2_kernel(const int* coors, const int* grid,
                             const _Float16* Q2, const _Float16* K2, const _Float16* V2,
                             const _Float16* frag, const float* pos2,
                             const float* in_b, const float* out_b,
                             float* canvas, int N) {
  __shared__ _Float16 oT[16 * RS];
  __shared__ int flat0s[16];
  int lane = threadIdx.x;
  int bidx = blockIdx.y;
  long base = (long)blockIdx.x * 16;
  const int sh0[9] = {0,-1,1,0,-1,1,0,-1, 1};
  const int sh1[9] = {0, 0,0,1, 1,1,-1,-1,-1};

  int m = lane >> 1, h = lane & 1;           // 2 lanes per voxel (lane = head)
  long vox = base + m; if (vox >= N) vox = N - 1;
  int c0 = coors[((size_t)bidx * N + vox) * 2 + 0];
  int c1 = coors[((size_t)bidx * N + vox) * 2 + 1];
  if (h == 0) flat0s[m] = c0 * Wc + c1;

  float qd[16];
  {
    const v8h* qr = (const v8h*)(Q2 + ((size_t)bidx * N + vox) * Cc + h * 16);
    v8h q0 = qr[0], q1 = qr[1];
#pragma unroll
    for (int d = 0; d < 8; ++d) { qd[d] = (float)q0[d]; qd[8 + d] = (float)q1[d]; }
  }

  const int* gb = grid + (size_t)bidx * Hc * Wc;
  const float* bk2 = in_b + Cc     + h * 16; // invalid slots keep in_w biases
  const float* bv2 = in_b + 2 * Cc + h * 16;
  int idxs[9]; float logit[9];
#pragma unroll
  for (int s = 0; s < 9; ++s) {
    int r = c0 + sh0[s], cl = c1 + sh1[s];
    bool val = (r >= 0) & (r < Hc) & (cl >= 0) & (cl < Wc);
    int idx = val ? gb[r * Wc + cl] : -1;
    idxs[s] = idx;
    float d = 0.f;
    if (idx >= 0) {
      const v8h* kr = (const v8h*)(K2 + ((size_t)bidx * N + idx) * Cc + h * 16);
      v8h k0 = kr[0], k1 = kr[1];
#pragma unroll
      for (int dd = 0; dd < 8; ++dd) {
        d = fmaf((float)k0[dd], qd[dd], d);
        d = fmaf((float)k1[dd], qd[8 + dd], d);
      }
    } else {
#pragma unroll
      for (int dd = 0; dd < 16; ++dd) d = fmaf(bk2[dd], qd[dd], d);
    }
    logit[s] = d * 0.25f;                    // 1/sqrt(HD=16)
  }
  float mx = logit[0];
#pragma unroll
  for (int s = 1; s < 9; ++s) mx = fmaxf(mx, logit[s]);
  float p[9], sum = 0.f;
#pragma unroll
  for (int s = 0; s < 9; ++s) { p[s] = __expf(logit[s] - mx); sum += p[s]; }
  float inv = 1.f / sum;

  float o[16];
#pragma unroll
  for (int dd = 0; dd < 16; ++dd) o[dd] = 0.f;
#pragma unroll
  for (int s = 0; s < 9; ++s) {
    float ps = p[s] * inv;
    if (idxs[s] >= 0) {
      const v8h* vr = (const v8h*)(V2 + ((size_t)bidx * N + idxs[s]) * Cc + h * 16);
      v8h v0 = vr[0], v1 = vr[1];
      const float* p2 = pos2 + s * Cc + h * 16;
#pragma unroll
      for (int dd = 0; dd < 8; ++dd) {
        o[dd]     = fmaf(ps, (float)v0[dd] + p2[dd],     o[dd]);
        o[8 + dd] = fmaf(ps, (float)v1[dd] + p2[8 + dd], o[8 + dd]);
      }
    } else {
#pragma unroll
      for (int dd = 0; dd < 16; ++dd) o[dd] = fmaf(ps, bv2[dd], o[dd]);
    }
  }
  {
    v8h a0, a1;
#pragma unroll
    for (int dd = 0; dd < 8; ++dd) { a0[dd] = (_Float16)o[dd]; a1[dd] = (_Float16)o[8 + dd]; }
    v8h* dst = (v8h*)(oT + m * RS + h * 16);
    dst[0] = a0; dst[1] = a1;                // 2x ds_store_b128
  }
  __syncthreads();

  // out projection: D = out_w x o^T ; lane = voxel, VGPR = channel
  v16h bact = load_actB(oT, lane);
  int hi2 = lane >> 4, lo2 = lane & 15;
  int f0 = flat0s[lo2];
  bool ok = base + lo2 < (long)N;
#pragma unroll
  for (int nh = 0; nh < 2; ++nh) {
    v16h a = load_frag(frag, 6 + nh, lane);
    const float* bb = out_b + nh * 16 + 8 * hi2;
    v8f c;
#pragma unroll
    for (int r = 0; r < 8; ++r) c[r] = bb[r];
    c = __builtin_amdgcn_wmma_f32_16x16x32_f16(false, a, false, bact, (short)0, c, false, false);
    if (ok) {
#pragma unroll
      for (int r = 0; r < 8; ++r) {
        int n = nh * 16 + 8 * hi2 + r;
        canvas[((size_t)bidx * Cc + n) * (Hc * Wc) + f0] = c[r];
      }
    }
  }
}

extern "C" void kernel_launch(void* const* d_in, const int* in_sizes, int n_in,
                              void* d_out, int out_size, void* d_ws, size_t ws_size,
                              hipStream_t stream) {
  const float* feats = (const float*)d_in[0];
  const int*   coors = (const int*)d_in[1];
  const float* ln_w  = (const float*)d_in[2];
  const float* ln_b  = (const float*)d_in[3];
  const float* wq    = (const float*)d_in[4];
  const float* bq    = (const float*)d_in[5];
  const float* wk    = (const float*)d_in[6];
  const float* bk    = (const float*)d_in[7];
  const float* wv    = (const float*)d_in[8];
  const float* bv    = (const float*)d_in[9];
  const float* pos_w = (const float*)d_in[10];
  const float* pos_b = (const float*)d_in[11];
  const float* in_w  = (const float*)d_in[12];
  const float* in_b  = (const float*)d_in[13];
  const float* out_w = (const float*)d_in[14];
  const float* out_b = (const float*)d_in[15];
  float* canvas = (float*)d_out;
  int N = in_sizes[0] / (Bn * Cc);           // 120000

  // ws: [grid 2MB][frag 8KB][beff .5KB][pos2 1.25KB][Q2|K2|V2 f16 B*N*C each]
  char* ws = (char*)d_ws;
  int* grid = (int*)ws;
  size_t off = (size_t)Bn * Hc * Wc * sizeof(int);            // 2097152
  _Float16* frag = (_Float16*)(ws + off); off += 8 * 32 * 16 * sizeof(_Float16);
  float* beff = (float*)(ws + off);        off += 512;
  float* pos2 = (float*)(ws + off);        off += 1280;
  _Float16* Q2 = (_Float16*)(ws + off);
  _Float16* K2 = Q2 + (size_t)Bn * N * Cc;
  _Float16* V2 = K2 + (size_t)Bn * N * Cc;

  long canvas4 = (long)Bn * Cc * Hc * Wc / 4;
  int gridN = Bn * Hc * Wc;
  hipLaunchKernelGGL(init_kernel, dim3(2048), dim3(256), 0, stream,
                     canvas, canvas4, grid, gridN);
  long tot = (long)Bn * N;
  hipLaunchKernelGGL(scatter_grid_kernel, dim3((unsigned)((tot + 255) / 256)), dim3(256),
                     0, stream, coors, grid, N);
  hipLaunchKernelGGL(prep_kernel, dim3(1), dim3(1024), 0, stream,
                     wq, bq, wk, bk, wv, bv, in_w, in_b, out_w, pos_w, pos_b,
                     frag, beff, pos2);
  int tiles = (N + 15) / 16;                 // 7500, exact
  dim3 g(tiles, Bn);
  hipLaunchKernelGGL(pass1_kernel, g, dim3(32), 0, stream,
                     feats, ln_w, ln_b, frag, beff, Q2, K2, V2, N);
  hipLaunchKernelGGL(pass2_kernel, g, dim3(32), 0, stream,
                     coors, grid, Q2, K2, V2, frag, pos2, in_b, out_b, canvas, N);
}
